// GatedSwitchGNN_62268435858151
// MI455X (gfx1250) — compile-verified, matrix operand order checked by reading, not run on hardware
//
#include <hip/hip_runtime.h>
#include <hip/hip_bf16.h>

// ---------------- CDNA5 WMMA types ----------------
typedef __attribute__((ext_vector_type(16))) __bf16 v16bf;
typedef __attribute__((ext_vector_type(8)))  float  v8f;

union Frag16 {            // 16 bf16 = 8 dwords = one wmma A/B operand per lane
    v16bf v;
    unsigned int   u[8];
    unsigned short s[16];
};

__device__ __forceinline__ unsigned short f2bf(float x) {
    union { float f; unsigned u; } c; c.f = x;
    unsigned u = c.u;
    unsigned r = u + 0x7fffu + ((u >> 16) & 1u);   // round-to-nearest-even
    return (unsigned short)(r >> 16);
}
__device__ __forceinline__ float bf2f(unsigned short h) {
    union { unsigned u; float f; } c; c.u = ((unsigned)h) << 16;
    return c.f;
}
__device__ __forceinline__ float sigmoidf(float x) {
    return 1.0f / (1.0f + __expf(-x));
}

// ---------------- problem constants ----------------
#define BB   8
#define VV   256
#define HIN  32
#define HH   64
#define EE   256
#define NSW  64
#define MM   320
#define V_LOW 0.81f
#define V_UPP 1.21f
#define EPS   1e-6f

// ---------------- workspace layout (bytes) ----------------
constexpr size_t SZ_S    = (size_t)BB*VV*VV*HH*2;          // 67,108,864 (bf16)
constexpr size_t OFF_SA  = 0;
constexpr size_t OFF_SB  = OFF_SA + SZ_S;
constexpr size_t OFF_HA  = OFF_SB + SZ_S;                  // f32 2048*64
constexpr size_t OFF_HB  = OFF_HA + 524288;
constexpr size_t OFF_HBFA= OFF_HB + 524288;                // bf16 2048*64
constexpr size_t OFF_HBFB= OFF_HBFA + 262144;
constexpr size_t OFF_XBF = OFF_HBFB + 262144;              // bf16 2048*32
constexpr size_t OFF_UH  = OFF_XBF + 131072;
constexpr size_t OFF_VHW = OFF_UH + 524288;
constexpr size_t OFF_WHW = OFF_VHW + 524288;
constexpr size_t OFF_BHW = OFF_WHW + 524288;
constexpr size_t OFF_CHW = OFF_BHW + 524288;
constexpr size_t OFF_R01 = OFF_CHW + 524288;               // 2*64 f32
constexpr size_t OFF_XG  = OFF_R01 + 1024;                 // 8*64 f32
constexpr size_t OFF_XGBF= OFF_XG + 2048;                  // 8*64 bf16
constexpr size_t OFF_XS  = OFF_XGBF + 1024;                // bf16 512*256
constexpr size_t OFF_HS1 = OFF_XS + 262144;                // f32 512*256
constexpr size_t OFF_SO  = OFF_HS1 + 524288;               // f32 512*4
constexpr size_t OFF_XC  = OFF_SO + 8192;                  // bf16 2048*192
constexpr size_t OFF_HC1 = OFF_XC + 786432;                // f32 2048*192
constexpr size_t OFF_CO  = OFF_HC1 + 1572864;              // f32 2048*3
constexpr size_t OFF_TOPO= OFF_CO + 24576;                 // f32 8*320 each below
constexpr size_t OFF_PFC = OFF_TOPO + 10240;
constexpr size_t OFF_VP  = OFF_PFC + 10240;
constexpr size_t OFF_VCH = OFF_VP + 10240;
constexpr size_t OFF_PAR = OFF_VCH + 10240;                // int
constexpr size_t OFF_CHD = OFF_PAR + 10240;                // int
constexpr size_t OFF_V   = OFF_CHD + 10240;                // f32 8*256
constexpr size_t OFF_QFC = OFF_V + 8192;
constexpr size_t OFF_PG  = OFF_QFC + 10240;
constexpr size_t OFF_QG  = OFF_PG + 8192;
// packed bf16 WMMA B-operands (fragment-major), reused stream-ordered
constexpr size_t OFF_PWN = OFF_QG + 8192;                  // 5 node weights, <=64x64 each
constexpr size_t OFF_PAW = OFF_PWN + 5*4096*2;             // Aw 64x64
constexpr size_t OFF_PMLP= OFF_PAW + 4096*2;               // max(256*256) bf16
constexpr size_t WS_END  = OFF_PMLP + 131072;

// ================= generic elementwise f32 -> bf16 =================
__global__ void k_f32_to_bf16(const float* __restrict__ in,
                              unsigned short* __restrict__ out, int n) {
    int i = blockIdx.x * blockDim.x + threadIdx.x;
    if (i < n) out[i] = f2bf(in[i]);
}

// ================= pack KxN f32 weight into bf16 B-fragment-major layout ====
// frag f = nt*(K/32) + kt ; element (f*32 + lane)*16 + t
// lane: col = lane&15 (N within tile), half = lane>>4 (K half), contiguous k = half*16 + t, same order as the A loads.
__global__ void k_pack_w(const float* __restrict__ W,
                         unsigned short* __restrict__ P, int Kdim, int Ndim) {
    int idx = blockIdx.x * blockDim.x + threadIdx.x;
    if (idx >= Kdim * Ndim) return;
    int t    = idx & 15;
    int lane = (idx >> 4) & 31;
    int f    = idx >> 9;
    int ktiles = Kdim >> 5;
    int kt = f % ktiles, nt = f / ktiles;
    int col = lane & 15, half = lane >> 4;
    int kk = kt*32 + half*16 + t;
    int n  = nt*16 + col;
    P[idx] = f2bf(W[(size_t)kk*Ndim + n]);
}

// ================= layer-0 collapsed einsum rows: r[sv] = W_emb[sv] @ Aw0 ====
__global__ void k_r01(const float* __restrict__ W_emb,    // (2,32)
                      const float* __restrict__ Aw0,      // (32,64)
                      float* __restrict__ r01) {          // (2,64)
    int t = threadIdx.x;                                  // 0..127
    if (t >= 128) return;
    int sv = t >> 6, k = t & 63;
    float acc = 0.f;
    #pragma unroll
    for (int h = 0; h < HIN; ++h) acc += W_emb[sv*HIN + h] * Aw0[h*HH + k];
    r01[sv*HH + k] = acc;
}

// ================= generic bf16 WMMA GEMM: C = act(A@Wp + bias) =============
// A: MxK bf16 row-major; Wp: packed fragment-major bf16; one wave per D tile.
__global__ __launch_bounds__(32)
void k_gemm_bf16(const unsigned short* __restrict__ A,
                 const unsigned short* __restrict__ Wp,
                 const float* __restrict__ bias,
                 float* __restrict__ C,
                 unsigned short* __restrict__ Cbf,
                 int Mdim, int Ndim, int Kdim, int do_relu) {
    const int nt = blockIdx.x, mt = blockIdx.y;
    const int lane = threadIdx.x, col = lane & 15, half = lane >> 4;
    const int n = nt * 16 + col;
    const int ktiles = Kdim >> 5;
    v8f acc = {};
    for (int kt = 0; kt < ktiles; ++kt) {
        Frag16 af, bf;
        const unsigned int* p32 =
            (const unsigned int*)(A + (size_t)(mt*16 + col)*Kdim + kt*32 + half*16);
        const unsigned int* q32 =
            (const unsigned int*)(Wp + ((size_t)(nt*ktiles + kt)*32 + lane)*16);
        #pragma unroll
        for (int t = 0; t < 8; ++t) { af.u[t] = p32[t]; bf.u[t] = q32[t]; }
        acc = __builtin_amdgcn_wmma_f32_16x16x32_bf16(
                  false, af.v, false, bf.v, (short)0, acc, false, false);
    }
    float bv = bias ? bias[n] : 0.f;
    #pragma unroll
    for (int r = 0; r < 8; ++r) {
        int m = mt*16 + r + half*8;
        float v = acc[r] + bv;
        if (do_relu) v = v > 0.f ? v : 0.f;
        C[(size_t)m*Ndim + n] = v;
        if (Cbf) Cbf[(size_t)m*Ndim + n] = f2bf(v);
    }
}

// ================= fused GatedGCN edge layer =================
// One wave per (b,i). Streams s over j-tiles, WMMA for e = s@Aw, fuses
// sigmoid gate, masked aggregations, residual, and h update. wave32 only.
__global__ __launch_bounds__(32)
void k_fused_layer(const unsigned short* __restrict__ s_in,   // bf16 (B,V,V,H)
                   unsigned short* __restrict__ s_out,        // bf16 (B,V,V,H)
                   const unsigned short* __restrict__ AwP,    // packed (2kt x 4nt)
                   const float* __restrict__ r01,             // (2,H) layer-0 path
                   const float* __restrict__ Smask,           // (B,V,V)
                   const float* __restrict__ Amask,           // (B,V,V)
                   const float* __restrict__ Uh, const float* __restrict__ Vh,
                   const float* __restrict__ Wh, const float* __restrict__ Bh,
                   const float* __restrict__ Ch,
                   const float* __restrict__ h_in,            // residual
                   float* __restrict__ h_out,
                   unsigned short* __restrict__ h_out_bf,
                   int has_s, int residual) {
    const int wave = blockIdx.x;                 // (b,i)
    const int b = wave >> 8, i = wave & 255;
    const int lane = threadIdx.x, col = lane & 15, half = lane >> 4;
    const size_t rowbase = ((size_t)b*VV + i)*VV;   // flat (b,i,j=0)

    // B fragments of Aw (K=64 -> 2 k-tiles, N=64 -> 4 n-tiles), from packed buf
    v16bf bB[2][4];
    if (has_s) {
        #pragma unroll
        for (int kt = 0; kt < 2; ++kt)
            #pragma unroll
            for (int nt = 0; nt < 4; ++nt) {
                Frag16 f;
                const unsigned int* q32 = (const unsigned int*)
                    (AwP + ((size_t)(nt*2 + kt)*32 + lane)*16);
                #pragma unroll
                for (int t = 0; t < 8; ++t) f.u[t] = q32[t];
                bB[kt][nt] = f.v;
            }
    }

    float bh[4];
    #pragma unroll
    for (int nt = 0; nt < 4; ++nt) bh[nt] = Bh[(size_t)wave*HH + nt*16 + col];

    float numA[4] = {0,0,0,0}, denA[4] = {0,0,0,0}, aggA[4] = {0,0,0,0};
    float degA = 0.f;

    for (int jt = 0; jt < 16; ++jt) {
        v8f acc[4] = {{}, {}, {}, {}};
        if (has_s) {
            if (jt < 15)   // prefetch next j-tile of the s stream
                __builtin_prefetch(s_in + (rowbase + (jt+1)*16 + col)*HH, 0, 0);
            #pragma unroll
            for (int kt = 0; kt < 2; ++kt) {
                Frag16 af;
                const unsigned int* p32 = (const unsigned int*)
                    (s_in + (rowbase + jt*16 + col)*HH + kt*32 + half*16);
                #pragma unroll
                for (int t = 0; t < 8; ++t) af.u[t] = p32[t];
                #pragma unroll
                for (int nt = 0; nt < 4; ++nt)
                    acc[nt] = __builtin_amdgcn_wmma_f32_16x16x32_bf16(
                        false, af.v, false, bB[kt][nt], (short)0, acc[nt],
                        false, false);
            }
        }
        #pragma unroll
        for (int nt = 0; nt < 4; ++nt) {
            const int h = nt*16 + col;
            #pragma unroll
            for (int r = 0; r < 8; ++r) {
                const int j = jt*16 + r + half*8;        // D layout: M = r + 8*half
                const size_t nodej = (size_t)b*VV + j;
                const float Sv = Smask[rowbase + j];
                const float chv = Ch[nodej*HH + h];
                float e;
                if (has_s) e = acc[nt][r] + bh[nt] + chv;
                else       e = r01[(Sv > 0.5f ? HH : 0) + h] + bh[nt] + chv;
                const float g = Sv * sigmoidf(e);
                float sv = e > 0.f ? e : 0.f;
                if (residual) sv += bf2f(s_in[(rowbase + j)*HH + h]);
                s_out[(rowbase + j)*HH + h] = f2bf(sv);
                numA[nt] += g * Wh[nodej*HH + h];
                denA[nt] += g;
                const float av = Amask[rowbase + j];
                aggA[nt] += av * Vh[nodej*HH + h];
                if (nt == 0) degA += av;
            }
        }
    }

    // combine the two half-wave j partitions (wave32)
    const float deg = degA + __shfl_xor(degA, 16, 32);
    #pragma unroll
    for (int nt = 0; nt < 4; ++nt) {
        const float num = numA[nt] + __shfl_xor(numA[nt], 16, 32);
        const float den = denA[nt] + __shfl_xor(denA[nt], 16, 32);
        const float agg = aggA[nt] + __shfl_xor(aggA[nt], 16, 32);
        if (half == 0) {
            const int h = nt*16 + col;
            float v = Uh[(size_t)wave*HH + h] + agg/(deg + EPS) + num/(den + EPS);
            v = v > 0.f ? v : 0.f;
            if (residual) v += h_in[(size_t)wave*HH + h];
            h_out[(size_t)wave*HH + h] = v;
            h_out_bf[(size_t)wave*HH + h] = f2bf(v);
        }
    }
}

// ================= graph embedding xg = sum_i h =================
__global__ void k_xg(const float* __restrict__ h, float* __restrict__ xg,
                     unsigned short* __restrict__ xgbf) {
    int b = blockIdx.x, t = threadIdx.x;      // 64 threads
    float acc = 0.f;
    for (int i = 0; i < VV; ++i) acc += h[((size_t)b*VV + i)*HH + t];
    xg[b*HH + t] = acc;
    xgbf[b*HH + t] = f2bf(acc);
}

// ================= switch-MLP input gather (bf16) =================
__global__ void k_xs_gather(const unsigned short* __restrict__ s_fin,
                            const unsigned short* __restrict__ h_fin,
                            const unsigned short* __restrict__ xgbf,
                            const int* __restrict__ sw_edges,
                            unsigned short* __restrict__ Xs) {
    int idx = blockIdx.x * blockDim.x + threadIdx.x;
    if (idx >= BB*NSW*256) return;
    int row = idx >> 8, c = idx & 255;
    int b = row / NSW, sw = row % NSW;
    int si = sw_edges[(b*NSW + sw)*2 + 0];
    int sj = sw_edges[(b*NSW + sw)*2 + 1];
    unsigned short v;
    if (c < 64)       v = s_fin[(((size_t)b*VV + si)*VV + sj)*HH + c];
    else if (c < 128) v = h_fin[((size_t)b*VV + si)*HH + (c - 64)];
    else if (c < 192) v = h_fin[((size_t)b*VV + sj)*HH + (c - 128)];
    else              v = xgbf[b*HH + (c - 192)];
    Xs[(size_t)row*256 + c] = v;
}

// ================= line-MLP input gather (bf16) =================
__global__ void k_xc_gather(const unsigned short* __restrict__ h_fin,
                            const unsigned short* __restrict__ xgbf,
                            const int* __restrict__ edges,
                            unsigned short* __restrict__ Xc) {
    int idx = blockIdx.x * blockDim.x + threadIdx.x;
    if (idx >= BB*EE*192) return;
    int row = idx / 192, c = idx % 192;
    int b = row / EE, e = row % EE;
    int ei = edges[(b*EE + e)*2 + 0];
    int ej = edges[(b*EE + e)*2 + 1];
    unsigned short v;
    if (c < 64)       v = h_fin[((size_t)b*VV + ei)*HH + c];
    else if (c < 128) v = h_fin[((size_t)b*VV + ej)*HH + (c - 64)];
    else              v = xgbf[b*HH + (c - 128)];
    Xc[(size_t)row*192 + c] = v;
}

// ================= tiny second MLP layer + sigmoid =================
__global__ void k_mlp2(const float* __restrict__ H, const float* __restrict__ W2,
                       const float* __restrict__ b2, float* __restrict__ out,
                       int rows, int K, int NC) {
    int idx = blockIdx.x * blockDim.x + threadIdx.x;
    if (idx >= rows * NC) return;
    int row = idx / NC, c = idx % NC;
    float acc = b2[c];
    for (int k = 0; k < K; ++k) acc += H[(size_t)row*K + k] * W2[k*NC + c];
    out[(size_t)row*NC + c] = sigmoidf(acc);
}

// ================= LinDistFlow assembly =================
__global__ void k_asm1(const float* __restrict__ CO, const float* __restrict__ SO,
                       const int* __restrict__ edges, const int* __restrict__ sw_edges,
                       float* __restrict__ topo, float* __restrict__ pfc,
                       float* __restrict__ vp, float* __restrict__ vch,
                       int* __restrict__ par, int* __restrict__ chd) {
    int idx = blockIdx.x * blockDim.x + threadIdx.x;
    if (idx >= BB*MM) return;
    int b = idx / MM, m = idx % MM;
    float tp, pf, vpv, vcv; int p, c;
    if (m < EE) {
        const float* cr = CO + (size_t)(b*EE + m)*3;
        tp = 1.f; pf = cr[0] - 0.5f;
        vpv = V_LOW*(1.f - cr[1]) + V_UPP*cr[1];
        vcv = V_LOW*(1.f - cr[2]) + V_UPP*cr[2];
        p = edges[(b*EE + m)*2 + 0]; c = edges[(b*EE + m)*2 + 1];
    } else {
        int sw = m - EE;
        const float* sr = SO + (size_t)(b*NSW + sw)*4;
        tp = sr[0]; pf = sr[1] - 0.5f;
        vpv = V_LOW*(1.f - sr[2]) + V_UPP*sr[2];
        vcv = V_LOW*(1.f - sr[3]) + V_UPP*sr[3];
        p = sw_edges[(b*NSW + sw)*2 + 0]; c = sw_edges[(b*NSW + sw)*2 + 1];
    }
    topo[idx] = tp; pfc[idx] = pf * tp;
    vp[idx] = vpv; vch[idx] = vcv;
    par[idx] = p;  chd[idx] = c;
}

__global__ void k_init_v(const float* __restrict__ x_data,
                         float* __restrict__ v, float* __restrict__ pg,
                         float* __restrict__ qg) {
    int idx = blockIdx.x * blockDim.x + threadIdx.x;
    if (idx >= BB*VV) return;
    v[idx] = 0.f;
    pg[idx] = x_data[(size_t)idx*2 + 0];
    qg[idx] = x_data[(size_t)idx*2 + 1];
}

__global__ void k_scatter_v(const float* __restrict__ vp, const float* __restrict__ vch,
                            const int* __restrict__ par, const int* __restrict__ chd,
                            float* __restrict__ v) {
    int idx = blockIdx.x * blockDim.x + threadIdx.x;
    if (idx >= BB*MM) return;
    int b = idx / MM;
    atomicAdd(&v[b*VV + par[idx]], vp[idx]);
    atomicAdd(&v[b*VV + chd[idx]], vch[idx]);
}

__global__ void k_vfix(const float* __restrict__ inv_degree, float* __restrict__ v) {
    int idx = blockIdx.x * blockDim.x + threadIdx.x;
    if (idx >= BB*VV) return;
    int n = idx % VV;
    v[idx] = (n == 0) ? 1.0f : inv_degree[idx] * v[idx];
}

__global__ void k_qfc(const float* __restrict__ v, const float* __restrict__ pfc,
                      const float* __restrict__ Rall, const float* __restrict__ Xall,
                      const int* __restrict__ par, const int* __restrict__ chd,
                      float* __restrict__ qfc, float* __restrict__ pg,
                      float* __restrict__ qg) {
    int idx = blockIdx.x * blockDim.x + threadIdx.x;
    if (idx >= BB*MM) return;
    int b = idx / MM;
    int p = par[idx], c = chd[idx];
    float vd = v[b*VV + p] - v[b*VV + c];
    float pf = pfc[idx];
    float q = (vd - 2.f*Rall[idx]*pf) / (2.f*Xall[idx]);
    qfc[idx] = q;
    atomicAdd(&pg[b*VV + p],  pf); atomicAdd(&pg[b*VV + c], -pf);
    atomicAdd(&qg[b*VV + p],  q);  atomicAdd(&qg[b*VV + c], -q);
}

__global__ void k_out(const float* __restrict__ pfc, const float* __restrict__ v,
                      const float* __restrict__ topo, const float* __restrict__ qfc,
                      const float* __restrict__ pg, const float* __restrict__ qg,
                      float* __restrict__ out) {
    int idx = blockIdx.x * blockDim.x + threadIdx.x;
    const int ZS = 2*MM + VV;       // 896
    const int ZC = MM + 2*VV;       // 832
    if (idx >= BB*ZS + BB*ZC) return;
    if (idx < BB*ZS) {
        int b = idx / ZS, o = idx % ZS;
        float r;
        if (o < MM)            r = pfc[b*MM + o];
        else if (o < MM + VV)  r = v[b*VV + (o - MM)];
        else                   r = topo[b*MM + (o - MM - VV)];
        out[idx] = r;
    } else {
        int t = idx - BB*ZS;
        int b = t / ZC, o = t % ZC;
        float r;
        if (o < MM)            r = qfc[b*MM + o];
        else if (o < MM + VV)  r = pg[b*VV + (o - MM)];
        else                   r = qg[b*VV + (o - MM - VV)];
        out[idx] = r;
    }
}

// ================= host-side launch =================
extern "C" void kernel_launch(void* const* d_in, const int* in_sizes, int n_in,
                              void* d_out, int out_size, void* d_ws, size_t ws_size,
                              hipStream_t stream) {
    const float* x        = (const float*)d_in[0];
    const float* A        = (const float*)d_in[1];
    const float* S        = (const float*)d_in[2];
    const float* x_data   = (const float*)d_in[3];
    const float* inv_deg  = (const float*)d_in[4];
    const float* Rall     = (const float*)d_in[5];
    const float* Xall     = (const float*)d_in[6];
    const float* W_emb    = (const float*)d_in[10];
    const float* W0       = (const float*)d_in[11];   // (6,32,64)
    const float* Wl       = (const float*)d_in[12];   // (2,6,64,64)
    const float* smlp_W1  = (const float*)d_in[13];
    const float* smlp_b1  = (const float*)d_in[14];
    const float* smlp_W2  = (const float*)d_in[15];
    const float* smlp_b2  = (const float*)d_in[16];
    const float* cmlp_W1  = (const float*)d_in[17];
    const float* cmlp_b1  = (const float*)d_in[18];
    const float* cmlp_W2  = (const float*)d_in[19];
    const float* cmlp_b2  = (const float*)d_in[20];
    const int*   edges    = (const int*)d_in[21];
    const int*   sw_edges = (const int*)d_in[22];
    float* out = (float*)d_out;

    char* w = (char*)d_ws;
    unsigned short* SA   = (unsigned short*)(w + OFF_SA);
    unsigned short* SB   = (unsigned short*)(w + OFF_SB);
    float* HA            = (float*)(w + OFF_HA);
    float* HB            = (float*)(w + OFF_HB);
    unsigned short* HBFA = (unsigned short*)(w + OFF_HBFA);
    unsigned short* HBFB = (unsigned short*)(w + OFF_HBFB);
    unsigned short* XBF  = (unsigned short*)(w + OFF_XBF);
    float* UH  = (float*)(w + OFF_UH);
    float* VHW = (float*)(w + OFF_VHW);
    float* WHW = (float*)(w + OFF_WHW);
    float* BHW = (float*)(w + OFF_BHW);
    float* CHW = (float*)(w + OFF_CHW);
    float* R01 = (float*)(w + OFF_R01);
    float* XG  = (float*)(w + OFF_XG);
    unsigned short* XGBF = (unsigned short*)(w + OFF_XGBF);
    unsigned short* XS   = (unsigned short*)(w + OFF_XS);
    float* HS1 = (float*)(w + OFF_HS1);
    float* SO  = (float*)(w + OFF_SO);
    unsigned short* XC   = (unsigned short*)(w + OFF_XC);
    float* HC1 = (float*)(w + OFF_HC1);
    float* CO  = (float*)(w + OFF_CO);
    float* TOPO= (float*)(w + OFF_TOPO);
    float* PFC = (float*)(w + OFF_PFC);
    float* VP  = (float*)(w + OFF_VP);
    float* VCH = (float*)(w + OFF_VCH);
    int*   PAR = (int*)(w + OFF_PAR);
    int*   CHD = (int*)(w + OFF_CHD);
    float* VN  = (float*)(w + OFF_V);
    float* QFC = (float*)(w + OFF_QFC);
    float* PG  = (float*)(w + OFF_PG);
    float* QG  = (float*)(w + OFF_QG);
    unsigned short* PWN  = (unsigned short*)(w + OFF_PWN);  // 5 slots of 4096 bf16
    unsigned short* PAW  = (unsigned short*)(w + OFF_PAW);
    unsigned short* PMLP = (unsigned short*)(w + OFF_PMLP);

    // ---- encoder prep ----
    k_f32_to_bf16<<<(BB*VV*HIN + 255)/256, 256, 0, stream>>>(x, XBF, BB*VV*HIN);
    k_r01<<<1, 128, 0, stream>>>(W_emb, W0 + 3*HIN*HH, R01);

    // ---- layer 0 (HIN=32 -> H=64), residual=0, collapsed einsum ----
    const int wsel[5] = {0, 1, 2, 4, 5};           // U,V,W,Bw,Cw
    float* ntgt[5] = {UH, VHW, WHW, BHW, CHW};
    dim3 gN0(HH/16, BB*VV/16);
    for (int t = 0; t < 5; ++t) {
        k_pack_w<<<(HIN*HH + 255)/256, 256, 0, stream>>>(
            W0 + (size_t)wsel[t]*HIN*HH, PWN + t*4096, HIN, HH);
        k_gemm_bf16<<<gN0, 32, 0, stream>>>(XBF, PWN + t*4096, nullptr,
                                            ntgt[t], nullptr, BB*VV, HH, HIN, 0);
    }
    k_fused_layer<<<BB*VV, 32, 0, stream>>>(SB, SB, PAW, R01, S, A,
                                            UH, VHW, WHW, BHW, CHW, HA,
                                            HA, HBFA, /*has_s=*/0, /*residual=*/0);

    // ---- layers 1..2 (residual) ----
    for (int l = 0; l < 2; ++l) {
        const unsigned short* hbf_in = (l == 0) ? HBFA : HBFB;
        const float* h_in            = (l == 0) ? HA : HB;
        unsigned short* s_in         = (l == 0) ? SB : SA;
        unsigned short* s_out        = (l == 0) ? SA : SB;
        float* h_out                 = (l == 0) ? HB : HA;
        unsigned short* hbf_out      = (l == 0) ? HBFB : HBFA;
        const float* WL = Wl + (size_t)l*6*HH*HH;
        for (int t = 0; t < 5; ++t) {
            k_pack_w<<<(HH*HH + 255)/256, 256, 0, stream>>>(
                WL + (size_t)wsel[t]*HH*HH, PWN + t*4096, HH, HH);
            k_gemm_bf16<<<gN0, 32, 0, stream>>>(hbf_in, PWN + t*4096, nullptr,
                                                ntgt[t], nullptr, BB*VV, HH, HH, 0);
        }
        k_pack_w<<<(HH*HH + 255)/256, 256, 0, stream>>>(WL + 3*HH*HH, PAW, HH, HH);
        k_fused_layer<<<BB*VV, 32, 0, stream>>>(s_in, s_out, PAW, R01,
                                                S, A, UH, VHW, WHW, BHW, CHW,
                                                h_in, h_out, hbf_out, 1, 1);
    }
    // final: h = HA / HBFA, s = SB

    // ---- readout heads ----
    k_xg<<<BB, HH, 0, stream>>>(HA, XG, XGBF);
    k_xs_gather<<<(BB*NSW*256 + 255)/256, 256, 0, stream>>>(SB, HBFA, XGBF,
                                                            sw_edges, XS);
    k_pack_w<<<(256*256 + 255)/256, 256, 0, stream>>>(smlp_W1, PMLP, 256, 256);
    k_gemm_bf16<<<dim3(256/16, BB*NSW/16), 32, 0, stream>>>(XS, PMLP, smlp_b1,
                                                            HS1, nullptr,
                                                            BB*NSW, 256, 256, 1);
    k_mlp2<<<(BB*NSW*4 + 255)/256, 256, 0, stream>>>(HS1, smlp_W2, smlp_b2, SO,
                                                     BB*NSW, 256, 4);
    k_xc_gather<<<(BB*EE*192 + 255)/256, 256, 0, stream>>>(HBFA, XGBF, edges, XC);
    k_pack_w<<<(192*192 + 255)/256, 256, 0, stream>>>(cmlp_W1, PMLP, 192, 192);
    k_gemm_bf16<<<dim3(192/16, BB*EE/16), 32, 0, stream>>>(XC, PMLP, cmlp_b1,
                                                           HC1, nullptr,
                                                           BB*EE, 192, 192, 1);
    k_mlp2<<<(BB*EE*3 + 255)/256, 256, 0, stream>>>(HC1, cmlp_W2, cmlp_b2, CO,
                                                    BB*EE, 192, 3);

    // ---- LinDistFlow assembly (edge-sparse incidence) ----
    k_asm1<<<(BB*MM + 255)/256, 256, 0, stream>>>(CO, SO, edges, sw_edges,
                                                  TOPO, PFC, VP, VCH, PAR, CHD);
    k_init_v<<<(BB*VV + 255)/256, 256, 0, stream>>>(x_data, VN, PG, QG);
    k_scatter_v<<<(BB*MM + 255)/256, 256, 0, stream>>>(VP, VCH, PAR, CHD, VN);
    k_vfix<<<(BB*VV + 255)/256, 256, 0, stream>>>(inv_deg, VN);
    k_qfc<<<(BB*MM + 255)/256, 256, 0, stream>>>(VN, PFC, Rall, Xall, PAR, CHD,
                                                 QFC, PG, QG);
    k_out<<<(BB*(2*MM+VV) + BB*(MM+2*VV) + 255)/256, 256, 0, stream>>>(
        PFC, VN, TOPO, QFC, PG, QG, out);
}
